// Encoder_13211319403093
// MI455X (gfx1250) — compile-verified, matrix operand order checked by reading, not run on hardware
//
#include <hip/hip_runtime.h>
#include <hip/hip_bf16.h>

// ---------------------------------------------------------------------------
// 4-layer LSTM, B=256, T=100, NIN=64, H=1024.
// bf16 WMMA (v_wmma_f32_16x16x32_bf16), weights L2-resident after one
// fp32->bf16 conversion. Anti-diagonal schedule (one launch per d = l + t).
// Inner GEMM loop is software-pipelined with explicit double-buffered
// A/B fragments so loads for step k+1 overlap the WMMAs of step k.
// ---------------------------------------------------------------------------

#define BB   256      // batch
#define TT   100      // timesteps
#define NIN  64       // layer-1 input dim
#define HH   1024     // hidden
#define G4   4096     // 4*H gate columns

typedef __attribute__((ext_vector_type(16))) __bf16 v16bf;
typedef __attribute__((ext_vector_type(8)))  __bf16 v8bf;
typedef __attribute__((ext_vector_type(8)))  float  v8f;

// ---------------------------------------------------------------------------
__global__ void cvt_f32_bf16_kernel(const float* __restrict__ in,
                                    __bf16* __restrict__ out, int n) {
    int i = blockIdx.x * blockDim.x + threadIdx.x;
    if (i < n) out[i] = (__bf16)in[i];
}

__global__ void bias_combine_kernel(const float* __restrict__ a,
                                    const float* __restrict__ b,
                                    float* __restrict__ o, int n) {
    int i = blockIdx.x * blockDim.x + threadIdx.x;
    if (i < n) o[i] = a[i] + b[i];
}

// ---------------------------------------------------------------------------
struct LayerArgs {
    const __bf16* Xin;    // [256, xstride] layer input (bf16)
    const __bf16* Wih;    // [4096, din]  row-major
    const __bf16* Whh;    // [4096, 1024] row-major
    const __bf16* Hprev;  // [256, 1024]
    const float*  bias;   // [4096] (bih+bhh)
    float*        C;      // [256, 1024] in/out cell state
    __bf16*       Hout;   // [256, 1024] bf16 h for next GEMMs
    float*        OutF;   // fp32 h output (top layer) or null
    int xstride, din, outf_stride;
};
struct DiagArgs { LayerArgs e[4]; };

__device__ __forceinline__ v16bf load_frag(const __bf16* p) {
    // lane<16 : K {0..7} U {16..23}; lane>=16 : K {8..15} U {24..31}
    // caller has already added (8*half) to p.
    v8bf lo = *(const v8bf*)(p);
    v8bf hi = *(const v8bf*)(p + 16);
    return __builtin_shufflevector(lo, hi, 0, 1, 2, 3, 4, 5, 6, 7,
                                           8, 9, 10, 11, 12, 13, 14, 15);
}

__device__ __forceinline__ float sigmoidf_fast(float x) {
    return 1.0f / (1.0f + __expf(-x));
}

// Software-pipelined K-loop: fragments for k+32 are loaded (into a disjoint
// register set) before the 8 WMMAs of step k are issued, so each WMMA group
// waits only on the previous load group, not the one just issued.
__device__ __forceinline__ void gemm_pass(const __bf16* a_base,
                                          const __bf16* const* wp,
                                          int K, v8f* acc) {
    v16bf ac = load_frag(a_base);
    v16bf wc[8];
    #pragma unroll
    for (int i = 0; i < 8; ++i) wc[i] = load_frag(wp[i]);

    for (int k0 = 0; k0 + 32 < K; k0 += 32) {
        v16bf an = load_frag(a_base + k0 + 32);
        v16bf wn[8];
        #pragma unroll
        for (int i = 0; i < 8; ++i) wn[i] = load_frag(wp[i] + k0 + 32);
        #pragma unroll
        for (int i = 0; i < 8; ++i)
            acc[i] = __builtin_amdgcn_wmma_f32_16x16x32_bf16(
                false, ac, false, wc[i], (short)0, acc[i], false, false);
        ac = an;
        #pragma unroll
        for (int i = 0; i < 8; ++i) wc[i] = wn[i];
    }
    #pragma unroll
    for (int i = 0; i < 8; ++i)
        acc[i] = __builtin_amdgcn_wmma_f32_16x16x32_bf16(
            false, ac, false, wc[i], (short)0, acc[i], false, false);
}

// ---------------------------------------------------------------------------
// One diagonal: blockIdx.y selects the (layer, timestep) cell.
// Per cell: 512 waves = 16 batch-tiles x 32 column-strips (32 cols each).
// Per wave: 16x32 patch for each of the 4 gates => 8 f32 accumulators,
// fragment index i = gate*2 + colhalf.
// ---------------------------------------------------------------------------
__global__ __launch_bounds__(256) void lstm_diag_kernel(DiagArgs args) {
    const LayerArgs la = args.e[blockIdx.y];

    const int tid    = threadIdx.x;
    const int lane   = tid & 31;
    const int wave   = blockIdx.x * (blockDim.x >> 5) + (tid >> 5); // 0..511
    const int mtile  = wave >> 5;   // 0..15 (batch rows)
    const int cstrip = wave & 31;   // 0..31 (32-col strip within H)
    const int m0     = mtile * 16;
    const int c0     = cstrip * 32;

    const int half = lane >> 4;
    const int l15  = lane & 15;

    v8f acc[8];
    #pragma unroll
    for (int i = 0; i < 8; ++i) acc[i] = (v8f){};

    // ---- pass 1: Xin @ Wih^T  (K = din) ----
    {
        const int din = la.din;
        const __bf16* a_base = la.Xin + (size_t)(m0 + l15) * la.xstride + 8 * half;
        const __bf16* wp[8];
        #pragma unroll
        for (int g = 0; g < 4; ++g)
            #pragma unroll
            for (int j = 0; j < 2; ++j)
                wp[g * 2 + j] = la.Wih +
                    (size_t)(g * HH + c0 + 16 * j + l15) * din + 8 * half;
        gemm_pass(a_base, wp, din, acc);
    }

    // ---- pass 2: Hprev @ Whh^T  (K = 1024) ----
    {
        const __bf16* a_base = la.Hprev + (size_t)(m0 + l15) * HH + 8 * half;
        const __bf16* wp[8];
        #pragma unroll
        for (int g = 0; g < 4; ++g)
            #pragma unroll
            for (int j = 0; j < 2; ++j)
                wp[g * 2 + j] = la.Whh +
                    (size_t)(g * HH + c0 + 16 * j + l15) * HH + 8 * half;
        gemm_pass(a_base, wp, HH, acc);
    }

    // ---- fused epilogue: gate activations + state update ----
    // C/D layout: lane holds column N = l15 (within tile); rows M = 8*half + r
    #pragma unroll
    for (int j = 0; j < 2; ++j) {
        const int cc = c0 + 16 * j + l15;
        const float bi = la.bias[0 * HH + cc];
        const float bf = la.bias[1 * HH + cc];
        const float bg = la.bias[2 * HH + cc];
        const float bo = la.bias[3 * HH + cc];
        #pragma unroll
        for (int r = 0; r < 8; ++r) {
            const int m = m0 + 8 * half + r;
            const size_t idx = (size_t)m * HH + cc;
            float iv = sigmoidf_fast(acc[0 * 2 + j][r] + bi);
            float fv = sigmoidf_fast(acc[1 * 2 + j][r] + bf);
            float gv = tanhf(acc[2 * 2 + j][r] + bg);
            float ov = sigmoidf_fast(acc[3 * 2 + j][r] + bo);
            float cn = fv * la.C[idx] + iv * gv;
            la.C[idx] = cn;
            float hn = ov * tanhf(cn);
            la.Hout[idx] = (__bf16)hn;
            if (la.OutF) la.OutF[(size_t)m * la.outf_stride + cc] = hn;
        }
    }
}

// ---------------------------------------------------------------------------
// Host-side orchestration
// ---------------------------------------------------------------------------
static inline size_t align256(size_t x) { return (x + 255) & ~(size_t)255; }

extern "C" void kernel_launch(void* const* d_in, const int* in_sizes, int n_in,
                              void* d_out, int out_size, void* d_ws, size_t ws_size,
                              hipStream_t stream) {
    (void)in_sizes; (void)n_in; (void)out_size; (void)ws_size;

    const float* x = (const float*)d_in[0];
    const float* Wih[4]; const float* Whh[4];
    const float* bih[4]; const float* bhh[4];
    for (int l = 0; l < 4; ++l) {
        Wih[l] = (const float*)d_in[1 + 4 * l];
        Whh[l] = (const float*)d_in[2 + 4 * l];
        bih[l] = (const float*)d_in[3 + 4 * l];
        bhh[l] = (const float*)d_in[4 + 4 * l];
    }
    const int dins[4] = {NIN, HH, HH, HH};

    // ---- workspace layout (all 256B-aligned) ----
    char* ws = (char*)d_ws;
    size_t off = 0;
    __bf16* xbf = (__bf16*)(ws + off);
    off = align256(off + (size_t)BB * TT * NIN * sizeof(__bf16));
    __bf16* wihB[4]; __bf16* whhB[4];
    for (int l = 0; l < 4; ++l) {
        wihB[l] = (__bf16*)(ws + off);
        off = align256(off + (size_t)G4 * dins[l] * sizeof(__bf16));
        whhB[l] = (__bf16*)(ws + off);
        off = align256(off + (size_t)G4 * HH * sizeof(__bf16));
    }
    float* biasC[4];
    for (int l = 0; l < 4; ++l) {
        biasC[l] = (float*)(ws + off);
        off = align256(off + (size_t)G4 * sizeof(float));
    }
    __bf16* hbuf = (__bf16*)(ws + off);              // [4 layers][2 ping-pong][256][1024]
    off = align256(off + (size_t)4 * 2 * BB * HH * sizeof(__bf16));
    float* cbuf = (float*)(ws + off);                // layers 0..2 cell state
    off = align256(off + (size_t)3 * BB * HH * sizeof(float));

    float* out = (float*)d_out;                      // [256,100,1024]
    float* c4  = out + (size_t)BB * TT * HH;         // layer-4 cell state lives here

    // ---- one-time (per launch) conversions ----
    {
        int n = BB * TT * NIN;
        cvt_f32_bf16_kernel<<<(n + 255) / 256, 256, 0, stream>>>(x, xbf, n);
        for (int l = 0; l < 4; ++l) {
            int nih = G4 * dins[l];
            cvt_f32_bf16_kernel<<<(nih + 255) / 256, 256, 0, stream>>>(Wih[l], wihB[l], nih);
            int nhh = G4 * HH;
            cvt_f32_bf16_kernel<<<(nhh + 255) / 256, 256, 0, stream>>>(Whh[l], whhB[l], nhh);
            bias_combine_kernel<<<(G4 + 255) / 256, 256, 0, stream>>>(bih[l], bhh[l], biasC[l], G4);
        }
        hipMemsetAsync(hbuf, 0, (size_t)4 * 2 * BB * HH * sizeof(__bf16), stream);
        hipMemsetAsync(cbuf, 0, (size_t)3 * BB * HH * sizeof(float), stream);
        hipMemsetAsync(c4,   0, (size_t)BB * HH * sizeof(float), stream);
    }

    // ---- wavefront schedule: diagonal d = l + t, one launch per diagonal ----
    for (int d = 0; d < TT + 3; ++d) {
        DiagArgs da;
        int n = 0;
        for (int l = 0; l < 4; ++l) {
            const int t = d - l;
            if (t < 0 || t >= TT) continue;
            const int rb = t & 1, wb = (t + 1) & 1;
            LayerArgs& e = da.e[n++];
            e.Xin = (l == 0)
                ? (xbf + (size_t)t * NIN)                          // x[b][t][:], stride T*NIN
                : (hbuf + ((size_t)(l - 1) * 2 + wb) * BB * HH);   // lower layer h @ t (prev diag)
            e.xstride = (l == 0) ? (TT * NIN) : HH;
            e.din     = dins[l];
            e.Wih     = wihB[l];
            e.Whh     = whhB[l];
            e.Hprev   = hbuf + ((size_t)l * 2 + rb) * BB * HH;     // own h @ t-1 (prev diag)
            e.bias    = biasC[l];
            e.C       = (l == 3) ? c4 : (cbuf + (size_t)l * BB * HH);
            e.Hout    = hbuf + ((size_t)l * 2 + wb) * BB * HH;
            e.OutF    = (l == 3) ? (out + (size_t)t * HH) : nullptr;
            e.outf_stride = TT * HH;
        }
        // 512 waves per cell = 64 blocks x 8 waves; blockIdx.y = cell index
        lstm_diag_kernel<<<dim3(64, n), 256, 0, stream>>>(da);
    }
}
